// Transformer_15994458210514
// MI455X (gfx1250) — matrix operand order repaired and checked by reading, not compiled, not run on hardware
//
#include <hip/hip_runtime.h>

#define D_MODEL 256
#define NHEAD   8
#define DFF     2048
#define NQ      300
#define BZ      32
#define TSEQ    1024
#define DV      32

typedef __attribute__((ext_vector_type(16))) __bf16      v16bf;
typedef __attribute__((ext_vector_type(8)))  float       v8f;
typedef __attribute__((ext_vector_type(4)))  unsigned int u32x4;

union Frag {
  v16bf  v;
  __bf16 e[16];
  u32x4  q[2];
};

__device__ __forceinline__ v8f wmma_bf16(const Frag& a, const Frag& b, v8f c) {
  return __builtin_amdgcn_wmma_f32_16x16x32_bf16(false, a.v, false, b.v, (short)0, c, false, false);
}

// ---------------- elementwise helpers ----------------

__global__ __launch_bounds__(256) void k_cvt_bf16(__bf16* __restrict__ dst,
                                                  const float* __restrict__ src, long n) {
  long i = (long)blockIdx.x * blockDim.x + threadIdx.x;
  if (i < n) dst[i] = (__bf16)src[i];
}

__global__ __launch_bounds__(256) void k_add(float* __restrict__ y,
                                             const float* __restrict__ x, long n) {
  long i = (long)blockIdx.x * blockDim.x + threadIdx.x;
  if (i < n) y[i] += x[i];
}

// cat[r, h*64 + j] = j<32 ? a[r, h*32+j] : b[r, h*32+j-32]  (per-head concat), bf16 out
__global__ __launch_bounds__(256) void k_interleave_bf16(const float* __restrict__ a,
                                                         const float* __restrict__ b,
                                                         __bf16* __restrict__ out, long rows) {
  long i = (long)blockIdx.x * blockDim.x + threadIdx.x;
  long total = rows * 512;
  if (i >= total) return;
  long r = i >> 9;
  int  c = (int)(i & 511);
  int  h = c >> 6, j = c & 63;
  float v = (j < 32) ? a[r * 256 + h * 32 + j] : b[r * 256 + h * 32 + (j - 32)];
  out[i] = (__bf16)v;
}

// ---------------- WMMA GEMM:  Y[M,N] = X[M,K] @ W[N,K]^T + bias ----------------
// One wave computes a 16x64 strip. A tile streamed via async-copy into double-buffered
// LDS (ASYNCcnt-gated); B fragments register-prefetched one K-step ahead.
// Requires M%16==0, N%64==0, K%64==0.

__global__ __launch_bounds__(256) void k_gemm_wmma(
    const __bf16* __restrict__ X, const __bf16* __restrict__ W,
    const float* __restrict__ bias,
    float* __restrict__ Yf, __bf16* __restrict__ Yb,
    int M, int N, int K, int accumulate, int relu)
{
  __shared__ __align__(16) unsigned char asmem[8][2][1024];  // per-wave double-buffered A tile

  int wib   = threadIdx.x >> 5;
  int wave  = blockIdx.x * (blockDim.x >> 5) + wib;
  int lane  = threadIdx.x & 31;
  int tilesM = M >> 4, tilesN = N >> 6;
  int tm = wave / tilesN, tn = wave - tm * tilesN;
  if (tm >= tilesM) return;

  int nl = lane & 15, hl = lane >> 4;
  int Ku = K >> 1;                          // row length in packed bf16 pairs (uints)
  const unsigned int* Xu = (const unsigned int*)X;
  const unsigned int* Wu = (const unsigned int*)W;

  long arow = (long)(tm * 16 + nl) * Ku;
  long wrow[4];
#pragma unroll
  for (int t = 0; t < 4; ++t) wrow[t] = (long)(tn * 64 + t * 16 + nl) * Ku;

  unsigned lb0 = (unsigned)(unsigned long long)&asmem[wib][0][0] + (unsigned)lane * 32u;
  unsigned lb1 = (unsigned)(unsigned long long)&asmem[wib][1][0] + (unsigned)lane * 32u;

  auto issueA = [&](int k0, unsigned lb) {
    const void* g0 = (const void*)(Xu + arow + k0 + 4 * hl);
    const void* g1 = (const void*)(Xu + arow + k0 + 8 + 4 * hl);
    asm volatile("global_load_async_to_lds_b128 %0, %1, off"
                 :: "v"(lb), "v"((unsigned long long)g0) : "memory");
    asm volatile("global_load_async_to_lds_b128 %0, %1, off"
                 :: "v"(lb + 16u), "v"((unsigned long long)g1) : "memory");
  };
  auto loadB = [&](Frag* d, int k0) {
#pragma unroll
    for (int t = 0; t < 4; ++t) {
      const u32x4* bp = (const u32x4*)(Wu + wrow[t] + k0 + 4 * hl);
      d[t].q[0] = bp[0];
      d[t].q[1] = bp[2];
    }
  };
  auto readA = [&](const unsigned char* buf) -> Frag {
    Frag a;
    const u32x4* ap = (const u32x4*)(buf + lane * 32);
    a.q[0] = ap[0];
    a.q[1] = ap[1];
    return a;
  };

  v8f acc[4];
#pragma unroll
  for (int t = 0; t < 4; ++t)
#pragma unroll
    for (int g = 0; g < 8; ++g) acc[t][g] = 0.0f;

  Frag b0[4], b1[4];
  issueA(0, lb0);
  loadB(b0, 0);

  int k0 = 0;
  while (true) {
    // even step: consume buf0/b0, prefetch into buf1/b1
    if (k0 + 16 < Ku) {
      issueA(k0 + 16, lb1);
      loadB(b1, k0 + 16);
      asm volatile("s_wait_asynccnt 0x2" ::: "memory");
    } else {
      asm volatile("s_wait_asynccnt 0x0" ::: "memory");
    }
    {
      Frag a = readA(&asmem[wib][0][0]);
#pragma unroll
      for (int t = 0; t < 4; ++t) acc[t] = wmma_bf16(a, b0[t], acc[t]);
    }
    k0 += 16; if (k0 >= Ku) break;

    // odd step: consume buf1/b1, prefetch into buf0/b0
    if (k0 + 16 < Ku) {
      issueA(k0 + 16, lb0);
      loadB(b0, k0 + 16);
      asm volatile("s_wait_asynccnt 0x2" ::: "memory");
    } else {
      asm volatile("s_wait_asynccnt 0x0" ::: "memory");
    }
    {
      Frag a = readA(&asmem[wib][1][0]);
#pragma unroll
      for (int t = 0; t < 4; ++t) acc[t] = wmma_bf16(a, b1[t], acc[t]);
    }
    k0 += 16; if (k0 >= Ku) break;
  }

#pragma unroll
  for (int t = 0; t < 4; ++t) {
    int col = tn * 64 + t * 16 + nl;
    float bv = bias ? bias[col] : 0.0f;
#pragma unroll
    for (int g = 0; g < 8; ++g) {
      int  row = tm * 16 + 8 * hl + g;       // C: lane=N, vgpr g -> M = g + 8*(lane/16)
      long idx = (long)row * N + col;
      float val = acc[t][g] + bv;
      if (accumulate) val += Yf[idx];
      if (relu) val = fmaxf(val, 0.0f);
      if (Yf) Yf[idx] = val;
      if (Yb) Yb[idx] = (__bf16)val;
    }
  }
}

// ---------------- Flash attention (one wave = one (b,h,16-row q tile)) ----------------
// Q [L,B,H*DH], K [S,B,H*DH], V [S,B,H*DV] all bf16; O [L,B,H*DV] f32.
// 32 S-columns per iteration; online softmax; P staged through LDS to A-layout.

template <int DH>
__global__ __launch_bounds__(128) void k_flash_attn(
    const __bf16* __restrict__ Q, const __bf16* __restrict__ Kx,
    const __bf16* __restrict__ V, float* __restrict__ O,
    int L, int S, float scale)
{
  constexpr int NF = DH / 32;
  constexpr int EQ = NHEAD * DH;
  constexpr int EV = NHEAD * DV;

  int wave = blockIdx.x * (blockDim.x >> 5) + (threadIdx.x >> 5);
  int lane = threadIdx.x & 31;
  int LT = (L + 15) >> 4;
  if (wave >= BZ * NHEAD * LT) return;

  int lt  = wave / (BZ * NHEAD);
  int rem = wave - lt * (BZ * NHEAD);
  int b = rem >> 3, h = rem & 7;
  int nl = lane & 15, hl = lane >> 4;
  int l0 = lt * 16;

  __shared__ __align__(16) __bf16 plds[4][16 * 32];
  __bf16* pl = plds[threadIdx.x >> 5];

  // Q fragments (A layout), packed b128 loads
  Frag qa[NF];
  {
    int row = l0 + nl; if (row >= L) row = L - 1;
    const __bf16* qr = Q + ((long)row * BZ + b) * EQ + h * DH;
#pragma unroll
    for (int f = 0; f < NF; ++f) {
      const u32x4* qp = (const u32x4*)(qr + f * 32 + 16 * hl);
      qa[f].q[0] = qp[0];
      qa[f].q[1] = qp[1];
    }
  }

  float mi[8], li[8];
  v8f oacc[2];
#pragma unroll
  for (int g = 0; g < 8; ++g) { mi[g] = -1e30f; li[g] = 0.0f; oacc[0][g] = 0.0f; oacc[1][g] = 0.0f; }

  for (int s0 = 0; s0 < S; s0 += 32) {
    v8f sc[2];
#pragma unroll
    for (int c = 0; c < 2; ++c) {
#pragma unroll
      for (int g = 0; g < 8; ++g) sc[c][g] = 0.0f;
      int srow = s0 + 16 * c + nl; if (srow >= S) srow = S - 1;
      const __bf16* kr = Kx + ((long)srow * BZ + b) * EQ + h * DH + 16 * hl;
#pragma unroll
      for (int f = 0; f < NF; ++f) {
        Frag kb;                              // B: lane = s-col, e -> d = f*32 + 16*hl + e
        const u32x4* kp4 = (const u32x4*)(kr + f * 32);
        kb.q[0] = kp4[0];
        kb.q[1] = kp4[1];
        sc[c] = wmma_bf16(qa[f], kb, sc[c]);
      }
      bool oob = (s0 + 16 * c + nl) >= S;
#pragma unroll
      for (int g = 0; g < 8; ++g) sc[c][g] = oob ? -1e30f : sc[c][g] * scale;
    }

    // online softmax per row (rows of a half live in that 16-lane half)
    float corr[8];
#pragma unroll
    for (int g = 0; g < 8; ++g) {
      float m01 = fmaxf(sc[0][g], sc[1][g]);
      for (int o = 1; o < 16; o <<= 1) m01 = fmaxf(m01, __shfl_xor(m01, o, 16));
      float mnew = fmaxf(mi[g], m01);
      corr[g] = __expf(mi[g] - mnew);
      mi[g] = mnew;
      float p0 = __expf(sc[0][g] - mnew);
      float p1 = __expf(sc[1][g] - mnew);
      sc[0][g] = p0; sc[1][g] = p1;
      float ps = p0 + p1;
      for (int o = 1; o < 16; o <<= 1) ps += __shfl_xor(ps, o, 16);
      li[g] = li[g] * corr[g] + ps;
      oacc[0][g] *= corr[g];
      oacc[1][g] *= corr[g];
    }

    // P: C-layout -> LDS -> A-layout (stores scattered b16, loads packed b128)
#pragma unroll
    for (int c = 0; c < 2; ++c)
#pragma unroll
      for (int g = 0; g < 8; ++g)
        pl[(g + 8 * hl) * 32 + 16 * c + nl] = (__bf16)sc[c][g];
    asm volatile("s_wait_dscnt 0" ::: "memory");
    Frag pa;
    {
      const u32x4* pp = (const u32x4*)(pl + nl * 32 + 16 * hl);
      pa.q[0] = pp[0];
      pa.q[1] = pp[1];
    }

    // V fragments (B layout; rows vary per element -> scalar bf16 gathers, no converts)
#pragma unroll
    for (int t = 0; t < 2; ++t) {
      Frag vb;
#pragma unroll
      for (int e = 0; e < 16; ++e) {
        int srow = s0 + 16 * hl + e; if (srow >= S) srow = S - 1;
        vb.e[e] = V[((long)srow * BZ + b) * EV + h * DV + 16 * t + nl];
      }
      oacc[t] = wmma_bf16(pa, vb, oacc[t]);
    }
  }

#pragma unroll
  for (int t = 0; t < 2; ++t)
#pragma unroll
    for (int g = 0; g < 8; ++g) {
      int row = l0 + 8 * hl + g;
      if (row < L)
        O[((long)row * BZ + b) * EV + h * DV + 16 * t + nl] = oacc[t][g] / li[g];
    }
}

// ---------------- residual + LayerNorm (one wave per 256-wide row) ----------------

__global__ __launch_bounds__(256) void k_res_ln(
    const float* __restrict__ X, const float* __restrict__ Dl,
    const float* __restrict__ G, const float* __restrict__ Bt,
    float* __restrict__ Y, __bf16* __restrict__ Yb, int rows)
{
  int wave = blockIdx.x * (blockDim.x >> 5) + (threadIdx.x >> 5);
  if (wave >= rows) return;
  int lane = threadIdx.x & 31;
  const float* xr = X  + (long)wave * D_MODEL;
  const float* dr = Dl + (long)wave * D_MODEL;
  float v[8]; float s = 0.0f;
#pragma unroll
  for (int i = 0; i < 8; ++i) { v[i] = xr[lane + 32 * i] + dr[lane + 32 * i]; s += v[i]; }
  for (int o = 1; o < 32; o <<= 1) s += __shfl_xor(s, o, 32);
  float mu = s * (1.0f / D_MODEL);
  float vs = 0.0f;
#pragma unroll
  for (int i = 0; i < 8; ++i) { float d = v[i] - mu; vs += d * d; }
  for (int o = 1; o < 32; o <<= 1) vs += __shfl_xor(vs, o, 32);
  float rstd = rsqrtf(vs * (1.0f / D_MODEL) + 1e-5f);
#pragma unroll
  for (int i = 0; i < 8; ++i) {
    int c = lane + 32 * i;
    float out = (v[i] - mu) * rstd * G[c] + Bt[c];
    if (Y)  Y [(long)wave * D_MODEL + c] = out;
    if (Yb) Yb[(long)wave * D_MODEL + c] = (__bf16)out;
  }
}

// ---------------- host orchestration ----------------

extern "C" void kernel_launch(void* const* d_in, const int* in_sizes, int n_in,
                              void* d_out, int out_size, void* d_ws, size_t ws_size,
                              hipStream_t stream)
{
  (void)in_sizes; (void)n_in; (void)out_size; (void)ws_size;
  const int  RB = 2 * NQ * BZ;   // 19200 boundary rows
  const int  RQ = NQ * BZ;       // 9600 per-side rows
  const int  RM = TSEQ * BZ;     // 32768 memory rows
  const long DD = (long)D_MODEL * D_MODEL;

  auto F = [&](int i) -> const float* { return (const float*)d_in[i]; };

  // --- input index map (setup_inputs insertion order; params (W,b) pairs) ---
  // 0..9 activations; 10..21 sa{qc,qp,kc,kp,v,out}; 22..35 ca_start{qc,qp,kc,kp,v,qsine,out};
  // 36..49 ca_end; 50/51 linear1; 52/53 linear2; 54..59 norm/norm1/norm2 (g,b).
  const int widx[22] = {10,12,14,16,18,20, 22,24,26,28,30,32,34, 36,38,40,42,44,46,48, 50,52};

  char* base = (char*)d_ws;
  size_t off = 0;
  auto alloc = [&](size_t bytes) -> char* {
    char* p = base + off; off = (off + bytes + 255) & ~(size_t)255; return p;
  };

  __bf16* wbf[22];
  for (int i = 0; i < 22; ++i) {
    long n = (i >= 20) ? (long)DFF * D_MODEL : DD;
    wbf[i] = (__bf16*)alloc(n * 2);
  }

  __bf16* bnd_bf     = (__bf16*)alloc((long)RB * D_MODEL * 2);
  __bf16* bndpos_bf  = (__bf16*)alloc((long)RB * D_MODEL * 2);
  float*  bndpost    = (float*) alloc((long)RB * D_MODEL * 4);
  __bf16* bndpost_bf = (__bf16*)alloc((long)RB * D_MODEL * 2);

  size_t scratch0 = off;
  // self-attn stage scratch
  float*  qS      = (float*) alloc((long)RB * D_MODEL * 4);
  float*  kS      = (float*) alloc((long)RB * D_MODEL * 4);
  float*  vS      = (float*) alloc((long)RB * D_MODEL * 4);
  __bf16* qS_bf   = (__bf16*)alloc((long)RB * D_MODEL * 2);
  __bf16* kS_bf   = (__bf16*)alloc((long)RB * D_MODEL * 2);
  __bf16* vS_bf   = (__bf16*)alloc((long)RB * D_MODEL * 2);
  float*  att1    = (float*) alloc((long)RB * D_MODEL * 4);
  __bf16* att1_bf = (__bf16*)alloc((long)RB * D_MODEL * 2);
  float*  delta1  = (float*) alloc((long)RB * D_MODEL * 4);
  // cross stage scratch (aliases the self scratch; stages are stream-ordered)
  off = scratch0;
  __bf16* mem_bf   = (__bf16*)alloc((long)RM * D_MODEL * 2);
  __bf16* pos_bf   = (__bf16*)alloc((long)RM * D_MODEL * 2);
  __bf16* qsine_bf = (__bf16*)alloc((long)RQ * D_MODEL * 2);
  float*  qq       = (float*) alloc((long)RQ * D_MODEL * 4);
  float*  qsb      = (float*) alloc((long)RQ * D_MODEL * 4);
  float*  kk       = (float*) alloc((long)RM * D_MODEL * 4);
  float*  kp       = (float*) alloc((long)RM * D_MODEL * 4);
  float*  vv       = (float*) alloc((long)RM * D_MODEL * 4);
  __bf16* vv_bf    = (__bf16*)alloc((long)RM * D_MODEL * 2);
  __bf16* qcat_bf  = (__bf16*)alloc((long)RQ * 512 * 2);
  __bf16* kcat_bf  = (__bf16*)alloc((long)RM * 512 * 2);
  float*  att2     = (float*) alloc((long)RQ * D_MODEL * 4);
  __bf16* att2_bf  = (__bf16*)alloc((long)RQ * D_MODEL * 2);
  float*  delta2   = (float*) alloc((long)RQ * D_MODEL * 4);
  float*  xf       = (float*) alloc((long)RQ * D_MODEL * 4);
  __bf16* xbf      = (__bf16*)alloc((long)RQ * D_MODEL * 2);
  __bf16* ffn_bf   = (__bf16*)alloc((long)RQ * DFF * 2);
  float*  ffn2     = (float*) alloc((long)RQ * D_MODEL * 4);

  auto CVT = [&](__bf16* dst, const float* src, long n) {
    k_cvt_bf16<<<dim3((unsigned)((n + 255) / 256)), dim3(256), 0, stream>>>(dst, src, n);
  };
  auto GEMM = [&](const __bf16* X, int wi, int bidx, float* Yf, __bf16* Yb,
                  int M, int N, int K, int acc, int relu) {
    long waves = (long)(M / 16) * (N / 64);
    k_gemm_wmma<<<dim3((unsigned)((waves + 7) / 8)), dim3(256), 0, stream>>>(
        X, wbf[wi], F(bidx), Yf, Yb, M, N, K, acc, relu);
  };

  // weight + activation bf16 conversions
  for (int i = 0; i < 22; ++i) {
    long n = (i >= 20) ? (long)DFF * D_MODEL : DD;
    CVT(wbf[i], F(widx[i]), n);
  }
  CVT(bnd_bf,                          F(0), (long)RQ * D_MODEL);
  CVT(bnd_bf    + (long)RQ * D_MODEL,  F(1), (long)RQ * D_MODEL);
  CVT(bndpos_bf,                       F(6), (long)RQ * D_MODEL);
  CVT(bndpos_bf + (long)RQ * D_MODEL,  F(7), (long)RQ * D_MODEL);

  // ---------- stage 1: joint self-attention ----------
  GEMM(bnd_bf,    0, 11, qS, nullptr, RB, 256, 256, 0, 0);
  GEMM(bndpos_bf, 1, 13, qS, qS_bf,   RB, 256, 256, 1, 0);
  GEMM(bnd_bf,    2, 15, kS, nullptr, RB, 256, 256, 0, 0);
  GEMM(bndpos_bf, 3, 17, kS, kS_bf,   RB, 256, 256, 1, 0);
  GEMM(bnd_bf,    4, 19, vS, vS_bf,   RB, 256, 256, 0, 0);
  {
    int L = 2 * NQ;
    long waves = (long)BZ * NHEAD * ((L + 15) / 16);
    k_flash_attn<32><<<dim3((unsigned)((waves + 3) / 4)), dim3(128), 0, stream>>>(
        qS_bf, kS_bf, vS_bf, att1, L, L, 0.17677669529663687f /* 1/sqrt(32) */);
  }
  CVT(att1_bf, att1, (long)RB * D_MODEL);
  GEMM(att1_bf, 5, 21, delta1, nullptr, RB, 256, 256, 0, 0);
  {
    unsigned lnB = (unsigned)((RQ + 7) / 8);
    k_res_ln<<<lnB, dim3(256), 0, stream>>>(F(0), delta1, F(54), F(55),
                                            bndpost, bndpost_bf, RQ);
    k_res_ln<<<lnB, dim3(256), 0, stream>>>(F(1), delta1 + (long)RQ * D_MODEL, F(54), F(55),
                                            bndpost + (long)RQ * D_MODEL,
                                            bndpost_bf + (long)RQ * D_MODEL, RQ);
  }

  // ---------- stage 2: per-side cross-attention + FFN ----------
  for (int s = 0; s < 2; ++s) {
    int wb = 6 + 7 * s;     // wbf index of ca_qcontent for this side
    int pb = 22 + 14 * s;   // d_in index of ca_qcontent.W for this side
    const __bf16* qpos_bf = bndpos_bf  + (long)s * RQ * D_MODEL;
    const __bf16* tgt_bf  = bndpost_bf + (long)s * RQ * D_MODEL;
    const float*  tgt_f   = bndpost    + (long)s * RQ * D_MODEL;

    CVT(mem_bf,   F(2 + s), (long)RM * D_MODEL);
    CVT(pos_bf,   F(4 + s), (long)RM * D_MODEL);
    CVT(qsine_bf, F(8 + s), (long)RQ * D_MODEL);

    GEMM(tgt_bf,   wb + 0, pb + 1,  qq,  nullptr, RQ, 256, 256, 0, 0);
    GEMM(qpos_bf,  wb + 1, pb + 3,  qq,  nullptr, RQ, 256, 256, 1, 0);
    GEMM(qsine_bf, wb + 5, pb + 11, qsb, nullptr, RQ, 256, 256, 0, 0);
    GEMM(pos_bf,   wb + 3, pb + 7,  kp,  nullptr, RM, 256, 256, 0, 0);
    GEMM(mem_bf,   wb + 2, pb + 5,  kk,  nullptr, RM, 256, 256, 0, 0);
    k_add<<<dim3((unsigned)(((long)RM * D_MODEL + 255) / 256)), dim3(256), 0, stream>>>(
        kk, kp, (long)RM * D_MODEL);
    GEMM(mem_bf,   wb + 4, pb + 9,  vv,  vv_bf, RM, 256, 256, 0, 0);

    k_interleave_bf16<<<dim3((unsigned)(((long)RQ * 512 + 255) / 256)), dim3(256), 0, stream>>>(
        qq, qsb, qcat_bf, RQ);
    k_interleave_bf16<<<dim3((unsigned)(((long)RM * 512 + 255) / 256)), dim3(256), 0, stream>>>(
        kk, kp, kcat_bf, RM);

    {
      long waves = (long)BZ * NHEAD * ((NQ + 15) / 16);
      k_flash_attn<64><<<dim3((unsigned)((waves + 3) / 4)), dim3(128), 0, stream>>>(
          qcat_bf, kcat_bf, vv_bf, att2, NQ, TSEQ, 0.125f /* 1/sqrt(64) */);
    }
    CVT(att2_bf, att2, (long)RQ * D_MODEL);
    GEMM(att2_bf, wb + 6, pb + 13, delta2, nullptr, RQ, 256, 256, 0, 0);

    unsigned lnB = (unsigned)((RQ + 7) / 8);
    k_res_ln<<<lnB, dim3(256), 0, stream>>>(tgt_f, delta2, F(56), F(57), xf, xbf, RQ);

    GEMM(xbf,    20, 51, nullptr, ffn_bf, RQ, DFF, 256, 0, 1);   // linear1 + ReLU -> bf16
    GEMM(ffn_bf, 21, 53, ffn2,   nullptr, RQ, 256, DFF, 0, 0);   // linear2

    k_res_ln<<<lnB, dim3(256), 0, stream>>>(xf, ffn2, F(58), F(59),
                                            (float*)d_out + (long)s * RQ * D_MODEL,
                                            nullptr, RQ);
  }
}
